// cINN_affine_54855322304897
// MI455X (gfx1250) — matrix operand order repaired and checked by reading, not compiled, not run on hardware
//
#include <hip/hip_runtime.h>
#include <hip/hip_bf16.h>

// cINN forward (12 affine coupling layers, 2x MLP(2->256->256x4->2) per layer)
// bf16 WMMA for the 256x256 hidden GEMMs. Each wave carries 32 batch rows
// (two 16-row M-tiles) through all 12 layers: every B fragment feeds two
// independent WMMA accumulation chains (2x arithmetic intensity on the weight
// stream, latency hiding inside one wave). All LDS traffic is wave-private;
// CDNA5 LDS is in-order per wave, so s_wait_dscnt + compiler fence suffices.

#define LNUM 12
#define H 256
#define NLH 4
#define ALPHA 0.3f
#define BATCH 131072

#define ROWS_PER_WAVE 32
#define WAVES_PER_BLOCK 2
#define BLOCK_THREADS (WAVES_PER_BLOCK * 32)
#define ROWS_PER_BLOCK (ROWS_PER_WAVE * WAVES_PER_BLOCK)

typedef __attribute__((ext_vector_type(16))) __bf16 bf16x16;
typedef __attribute__((ext_vector_type(8)))  __bf16 bf16x8;
typedef __attribute__((ext_vector_type(8)))  float  f32x8;

// Wave-local LDS fence: CDNA5 LDS ops are in-order within a wave, so a
// DScnt drain + compiler memory barrier is enough for cross-lane visibility.
#define LDS_WAVE_SYNC() asm volatile("s_wait_dscnt 0x0" ::: "memory")

struct alignas(16) WaveScratch {
  __bf16 hrow[ROWS_PER_WAVE][H + 8];   // activations, row-major bf16, +8 pad (bank spread)
  float  vbuf[ROWS_PER_WAVE][4];       // current 3-dim state per row
  float  u1buf[ROWS_PER_WAVE][2];      // gathered/padded u1 per row
  float  so[2][ROWS_PER_WAVE][2];      // net outputs: [0]=net A (pre-tanh s), [1]=net B (t)
};

// Coupling masks (j = layer % 6)
__device__ const int cM1[6][2]  = {{0,0},{1,0},{2,0},{0,1},{0,2},{1,2}};
__device__ const int cM1n[6]    = {1,1,1,2,2,2};
__device__ const int cM2[6][2]  = {{1,2},{0,2},{0,1},{2,0},{1,0},{0,0}};

// ---------------------------------------------------------------------------
// Prepack Wh (L,NL,H,H fp32, nets A+B) into bf16 WMMA B-fragment order:
// flat = mat*65536 + kb*8192 + nt*512 + lane*16 + e
//   N = nt*16 + (lane&15);  K = kb*32 + ((lane&16)?16:0) + e
// ---------------------------------------------------------------------------
__global__ void pack_wh_kernel(const float* __restrict__ A_Wh,
                               const float* __restrict__ B_Wh,
                               unsigned short* __restrict__ dst_u16) {
  const size_t idx = (size_t)blockIdx.x * blockDim.x + threadIdx.x;
  const size_t total = (size_t)2 * LNUM * NLH * H * H;
  if (idx >= total) return;
  const int e    = (int)(idx & 15);
  const int lane = (int)((idx >> 4) & 31);
  const int nt   = (int)((idx >> 9) & 15);
  const int kb   = (int)((idx >> 13) & 7);
  const int mat  = (int)(idx >> 16);          // 0..95
  const int net  = mat / (LNUM * NLH);        // 0=A, 1=B
  const int lm   = mat % (LNUM * NLH);        // layer*NLH + k
  const int N    = nt * 16 + (lane & 15);
  const int K    = kb * 32 + ((lane & 16) ? 16 : 0) + e;
  const float* src = net ? B_Wh : A_Wh;
  const float w = src[(size_t)lm * H * H + (size_t)K * H + N];
  __bf16* dst = (__bf16*)dst_u16;
  dst[idx] = (__bf16)w;
}

// ---------------------------------------------------------------------------
// One MLP: h = leaky(u1@W0+b0); 4x h = leaky(h@Wh+bh); out2 = h@Wf+bf
// Wave-private: 32 rows (two 16-row M-tiles) in wsv->hrow.
// ---------------------------------------------------------------------------
__device__ __forceinline__ void mlp_net(
    WaveScratch* wsv, const int lane,
    const float* __restrict__ W0, const float* __restrict__ b0,
    const __bf16* __restrict__ whp,          // packed fragments, 4 matrices
    const float* __restrict__ bh,
    const float* __restrict__ Wf, const float* __restrict__ bf,
    float* out2) {
  // ---- input layer: [32,2] @ [2,256] -> [32,256] (VALU, trivial) ----
  for (int idx = lane; idx < ROWS_PER_WAVE * H; idx += 32) {
    const int r = idx >> 8;
    const int c = idx & (H - 1);
    float x = wsv->u1buf[r][0] * W0[c] + wsv->u1buf[r][1] * W0[H + c] + b0[c];
    x = fmaxf(x, ALPHA * x);                 // LeakyReLU(0.3)
    wsv->hrow[r][c] = (__bf16)x;
  }

  // ---- 4 hidden layers: [32,256] @ [256,256] via v_wmma_f32_16x16x32_bf16 ----
  const int rloc  = lane & 15;
  const int khalf = (lane & 16) ? 16 : 0;    // A-layout K half select
  const int mbase = (lane & 16) ? 8 : 0;     // C-layout M base
  const int ncol  = lane & 15;               // C/B-layout N = lane%16
  for (int k = 0; k < NLH; ++k) {
    LDS_WAVE_SYNC();                         // prior hrow writes -> visible
    const __bf16* wmat = whp + (size_t)k * H * H;
    const float*  bias = bh + (size_t)k * H;

    // A fragments for both M-tiles (reused across all 16 n-tiles).
    // A k order used here (same as packed B): lanes 0-15 K=0..15, lanes 16-31 K=16..31.
    bf16x16 afrag[2][8];
#pragma unroll
    for (int mt = 0; mt < 2; ++mt) {
      const __bf16* hb = &wsv->hrow[mt * 16 + rloc][0];
#pragma unroll
      for (int kb = 0; kb < 8; ++kb) {
        bf16x8 lo = *(const bf16x8*)(hb + kb * 32 + khalf);        // ds_load_b128
        bf16x8 hi = *(const bf16x8*)(hb + kb * 32 + khalf + 8);    // ds_load_b128
        afrag[mt][kb] = __builtin_shufflevector(lo, hi,
            0,1,2,3,4,5,6,7,8,9,10,11,12,13,14,15);
      }
    }

    for (int nt = 0; nt < 16; ++nt) {
      const float bv = bias[nt * 16 + ncol];
      f32x8 acc0 = {bv, bv, bv, bv, bv, bv, bv, bv};
      f32x8 acc1 = {bv, bv, bv, bv, bv, bv, bv, bv};
#pragma unroll
      for (int kb = 0; kb < 8; ++kb) {
        const __bf16* fp = wmat + (((size_t)(kb * 16 + nt) * 32 + lane) * 16);
        bf16x8 blo = *(const bf16x8*)fp;        // global_load_b128 (L2/WGP$-hot)
        bf16x8 bhi = *(const bf16x8*)(fp + 8);  // global_load_b128
        bf16x16 bfrag = __builtin_shufflevector(blo, bhi,
            0,1,2,3,4,5,6,7,8,9,10,11,12,13,14,15);
        // Two independent accumulation chains share one B fragment.
        acc0 = __builtin_amdgcn_wmma_f32_16x16x32_bf16(
            false, afrag[0][kb], false, bfrag, (short)0, acc0, false, false);
        acc1 = __builtin_amdgcn_wmma_f32_16x16x32_bf16(
            false, afrag[1][kb], false, bfrag, (short)0, acc1, false, false);
      }
      // LeakyReLU + store C tiles back as bf16 (C layout: VGPR r -> M, lane -> N)
      const int col = nt * 16 + ncol;
#pragma unroll
      for (int rr = 0; rr < 8; ++rr) {
        float x0 = acc0[rr];
        x0 = fmaxf(x0, ALPHA * x0);
        wsv->hrow[mbase + rr][col] = (__bf16)x0;        // ds_store_b16
        float x1 = acc1[rr];
        x1 = fmaxf(x1, ALPHA * x1);
        wsv->hrow[16 + mbase + rr][col] = (__bf16)x1;   // ds_store_b16
      }
    }
  }

  LDS_WAVE_SYNC();
  // ---- output layer: [32,256] @ [256,2]; two (row,j) dots per lane ----
  {
    const int r  = lane >> 1;
    const int jo = lane & 1;
    float acc0 = bf[jo];
    float acc1 = bf[jo];
    const __bf16* hb0 = &wsv->hrow[r][0];
    const __bf16* hb1 = &wsv->hrow[r + 16][0];
    for (int kk = 0; kk < H; kk += 8) {
      bf16x8 h0 = *(const bf16x8*)(hb0 + kk);
      bf16x8 h1 = *(const bf16x8*)(hb1 + kk);
#pragma unroll
      for (int e = 0; e < 8; ++e) {
        const float w = Wf[(size_t)(kk + e) * 2 + jo];
        acc0 += (float)h0[e] * w;
        acc1 += (float)h1[e] * w;
      }
    }
    out2[r * 2 + jo] = acc0;
    out2[(r + 16) * 2 + jo] = acc1;
  }
}

// ---------------------------------------------------------------------------
// Main kernel: each wave carries 32 batch rows through all 12 coupling layers.
// One lane per row for the scalar coupling stages (no divergence).
// ---------------------------------------------------------------------------
__global__ __launch_bounds__(BLOCK_THREADS)
void cinn_forward_kernel(
    const float* __restrict__ u,
    const float* __restrict__ A_W0, const float* __restrict__ A_b0,
    const float* __restrict__ A_bh, const float* __restrict__ A_Wf,
    const float* __restrict__ A_bf,
    const float* __restrict__ B_W0, const float* __restrict__ B_b0,
    const float* __restrict__ B_bh, const float* __restrict__ B_Wf,
    const float* __restrict__ B_bf,
    const unsigned short* __restrict__ wpack_u16,
    float* __restrict__ out) {
  __shared__ WaveScratch smem[WAVES_PER_BLOCK];
  const int wave = threadIdx.x >> 5;
  const int lane = threadIdx.x & 31;
  WaveScratch* wsv = &smem[wave];
  const __bf16* wpack = (const __bf16*)wpack_u16;
  const int row0 = (blockIdx.x * WAVES_PER_BLOCK + wave) * ROWS_PER_WAVE;

  float ldj = 0.0f;
  {
    const float* up = u + (size_t)(row0 + lane) * 3;
    wsv->vbuf[lane][0] = up[0];
    wsv->vbuf[lane][1] = up[1];
    wsv->vbuf[lane][2] = up[2];
  }

  for (int li = LNUM - 1; li >= 0; --li) {   // direction = -1: reverse order
    const int j = li % 6;
    // gather u1 (padded to 2); one lane per row
    {
      const float va[3] = {wsv->vbuf[lane][0], wsv->vbuf[lane][1],
                           wsv->vbuf[lane][2]};
      wsv->u1buf[lane][0] = va[cM1[j][0]];
      wsv->u1buf[lane][1] = (cM1n[j] == 2) ? va[cM1[j][1]] : 0.0f;
    }
    LDS_WAVE_SYNC();

    // net A -> s (pre-tanh), net B -> t
    mlp_net(wsv, lane,
            A_W0 + (size_t)li * 2 * H, A_b0 + (size_t)li * H,
            wpack + (size_t)(0 * LNUM * NLH + li * NLH) * H * H,
            A_bh + (size_t)li * NLH * H,
            A_Wf + (size_t)li * H * 2, A_bf + (size_t)li * 2,
            &wsv->so[0][0][0]);
    mlp_net(wsv, lane,
            B_W0 + (size_t)li * 2 * H, B_b0 + (size_t)li * H,
            wpack + (size_t)(1 * LNUM * NLH + li * NLH) * H * H,
            B_bh + (size_t)li * NLH * H,
            B_Wf + (size_t)li * H * 2, B_bf + (size_t)li * 2,
            &wsv->so[1][0][0]);
    LDS_WAVE_SYNC();

    // coupling tail: s=tanh(.)*v2m, t=.*v2m, v2=exp(s)*u2+t, scatter, ldj+=sum(s)
    {
      const float va[3] = {wsv->vbuf[lane][0], wsv->vbuf[lane][1],
                           wsv->vbuf[lane][2]};
      float nv[3] = {va[0], va[1], va[2]};   // MASK1 positions keep old values
      const int n2 = 3 - cM1n[j];
      const float sa0 = wsv->so[0][lane][0];
      const float sa1 = wsv->so[0][lane][1];
      const float tb0 = wsv->so[1][lane][0];
      const float tb1 = wsv->so[1][lane][1];
      const int i20 = cM2[j][0];
      const float s0 = tanhf(sa0);
      nv[i20] = expf(s0) * va[i20] + tb0;
      float ls = s0;
      if (n2 == 2) {
        const int i21 = cM2[j][1];
        const float s1 = tanhf(sa1);
        nv[i21] = expf(s1) * va[i21] + tb1;
        ls += s1;
      }
      ldj += ls;
      wsv->vbuf[lane][0] = nv[0];
      wsv->vbuf[lane][1] = nv[1];
      wsv->vbuf[lane][2] = nv[2];
    }
    LDS_WAVE_SYNC();
  }

  {
    const size_t r = (size_t)(row0 + lane);
    out[r * 3 + 0] = wsv->vbuf[lane][0];
    out[r * 3 + 1] = wsv->vbuf[lane][1];
    out[r * 3 + 2] = wsv->vbuf[lane][2];
    out[(size_t)BATCH * 3 + r] = ldj;        // log_detJ appended after v
  }
}

// ---------------------------------------------------------------------------
extern "C" void kernel_launch(void* const* d_in, const int* in_sizes, int n_in,
                              void* d_out, int out_size, void* d_ws,
                              size_t ws_size, hipStream_t stream) {
  (void)in_sizes; (void)n_in; (void)out_size; (void)ws_size;
  const float* u    = (const float*)d_in[0];
  const float* A_W0 = (const float*)d_in[1];
  const float* A_b0 = (const float*)d_in[2];
  const float* A_Wh = (const float*)d_in[3];
  const float* A_bh = (const float*)d_in[4];
  const float* A_Wf = (const float*)d_in[5];
  const float* A_bf = (const float*)d_in[6];
  const float* B_W0 = (const float*)d_in[7];
  const float* B_b0 = (const float*)d_in[8];
  const float* B_Wh = (const float*)d_in[9];
  const float* B_bh = (const float*)d_in[10];
  const float* B_Wf = (const float*)d_in[11];
  const float* B_bf = (const float*)d_in[12];

  unsigned short* wpack = (unsigned short*)d_ws;  // 96 * 64K bf16 = 12.6 MB

  const size_t pack_total = (size_t)2 * LNUM * NLH * H * H;
  const int pack_blocks = (int)((pack_total + 255) / 256);
  pack_wh_kernel<<<pack_blocks, 256, 0, stream>>>(A_Wh, B_Wh, wpack);

  const int grid = BATCH / ROWS_PER_BLOCK;        // 2048 blocks of 64 threads
  cinn_forward_kernel<<<grid, BLOCK_THREADS, 0, stream>>>(
      u, A_W0, A_b0, A_bh, A_Wf, A_bf,
      B_W0, B_b0, B_bh, B_Wf, B_bf,
      wpack, (float*)d_out);
}
